// WindowAttention3D_764504179102
// MI455X (gfx1250) — compile-verified, matrix operand order checked by reading, not run on hardware
//
#include <hip/hip_runtime.h>
#include <hip/hip_bf16.h>
#include <stdint.h>

// ---------------------------------------------------------------------------
// CDNA5 WMMA helpers (gfx1250, wave32)
// ---------------------------------------------------------------------------
typedef __attribute__((ext_vector_type(16))) _Float16     v16h;
typedef __attribute__((ext_vector_type(8)))  float        v8f;
typedef __attribute__((ext_vector_type(4)))  unsigned int v4u;
typedef __attribute__((ext_vector_type(8)))  int          v8i;
typedef __attribute__((ext_vector_type(4)))  int          v4i;

#define WMMA_F32_F16(acc, a, b)                                                 \
  acc = __builtin_amdgcn_wmma_f32_16x16x32_f16(false, a, false, b, (short)0,    \
                                               acc, false, false)

// 16-bit A-fragment mapping (ISA 7.12.2, 16-bit A 16x32):
//   lane L: row = L&15, kgroup = L>>4
//   half h (0..15): K = (h&7) + ((h>>3)<<4) + (kgroup<<3)
// B fragment assumed to mirror this with N in place of M.
__device__ __forceinline__ int frag_k(int h, int kg) {
  return (h & 7) + ((h >> 3) << 4) + (kg << 3);
}

// Problem constants
#define NWIN   98                     // tokens per window (2*7*7)
#define CDIM   128
#define HEADS  4
#define HD     32
#define ROWS   (4096 * 98)            // 401408 token rows
#define QSCALE 0.17677669529663687f   // 32^-0.5

// ---------------------------------------------------------------------------
// Tensor Data Mover: 2D f16 tile load, global -> LDS, with optional LDS row
// padding (16 DWORDs data + 4 DWORDs pad -> 40-half LDS row stride) and
// zero-fill of out-of-bounds rows (tile_dim1 > tensor_dim1).
// D# bitfields per CDNA5 ISA §8.3/§8.4.
// ---------------------------------------------------------------------------
__device__ __forceinline__ void tdm_load_2d_f16(unsigned lds_byte_addr,
                                                const void* gaddr,
                                                unsigned row_halves,   // tile/tensor dim0
                                                unsigned valid_rows,   // tensor dim1
                                                unsigned tile_rows,    // tile dim1 (padded)
                                                unsigned row_stride_halves,
                                                bool pad_16dw_plus_4dw) {
  const unsigned long long ga = (unsigned long long)(uintptr_t)gaddr;
  v4u g0;
  g0[0] = 1u;                                   // count=1, is_restore=0
  g0[1] = lds_byte_addr;                        // lds_addr [63:32]
  g0[2] = (unsigned)(ga & 0xFFFFFFFFu);         // global_addr [95:64]
  g0[3] = (unsigned)((ga >> 32) & 0x01FFFFFFu)  // global_addr [120:96]
          | (2u << 30);                         // type=2 ("image")
  v8i g1;
  unsigned w0 = (1u << 16);                     // data_size = 1 -> 2 bytes
  if (pad_16dw_plus_4dw)
    w0 |= (1u << 20)                            // pad_enable
        | (3u << 22)                            // pad_interval: 16 DWORDs
        | (3u << 25);                           // pad_amount:   4 DWORDs
  g1[0] = (int)w0;                              // workgroup_mask=0 (no cluster)
  g1[1] = (int)(row_halves << 16);              // tensor_dim0 [79:48]
  g1[2] = (int)((valid_rows & 0xFFFFu) << 16);  // tensor_dim1 [111:80]
  g1[3] = (int)(row_halves << 16);              // tile_dim0   [127:112]
  g1[4] = (int)(tile_rows & 0xFFFFu);           // tile_dim1   [143:128]
  g1[5] = (int)row_stride_halves;               // tensor_dim0_stride [207:160]
  g1[6] = 0;
  g1[7] = 0;                                    // tensor_dim1_stride unused (2D)
  const v4i z4 = {0, 0, 0, 0};                  // groups 2/3 unused for 2D
#if __clang_major__ >= 23
  const v8i z8 = {0, 0, 0, 0, 0, 0, 0, 0};
  __builtin_amdgcn_tensor_load_to_lds(g0, g1, z4, z4, z8, 0);
#else
  __builtin_amdgcn_tensor_load_to_lds(g0, g1, z4, z4, 0);
#endif
}

// ---------------------------------------------------------------------------
// Kernel 0: one-shot f32 -> f16 conversion of the two weight matrices.
// ---------------------------------------------------------------------------
__global__ __launch_bounds__(256)
void convert_weights(const float* __restrict__ qkv_w,
                     const float* __restrict__ proj_w,
                     _Float16* __restrict__ wq, _Float16* __restrict__ wp) {
  const int i = blockIdx.x * 256 + threadIdx.x;
  if (i < 384 * 128) wq[i] = (_Float16)qkv_w[i];
  if (i < 128 * 128) wp[i] = (_Float16)proj_w[i];
}

// ---------------------------------------------------------------------------
// Kernel 1: qkv = x @ qkv_w^T  (401408 x 128) @ (128 x 384), f16 out,
//           q pre-scaled by hd^-0.5. 64-row blocks, 8 waves x 48 columns.
// ---------------------------------------------------------------------------
__global__ __launch_bounds__(256)
void qkv_gemm(const float* __restrict__ x, const _Float16* __restrict__ wq,
              _Float16* __restrict__ qkvh) {
  __shared__ _Float16 As[64 * 136];
  const int m0  = blockIdx.x * 64;
  const int tid = threadIdx.x;
  for (int i = tid; i < 64 * 128; i += 256) {
    int r = i >> 7, c = i & 127;
    As[r * 136 + c] = (_Float16)x[(size_t)(m0 + r) * 128 + c];
  }
  __syncthreads();

  const int wid = tid >> 5, lane = tid & 31;
  const int M = lane & 15, kg = lane >> 4;
  for (int ntl = 0; ntl < 3; ++ntl) {
    const int j = wid * 48 + ntl * 16 + M;      // output column this lane owns
    v16h bf[4];                                 // B fragments, reused over mt
#pragma unroll
    for (int kk = 0; kk < 4; ++kk)
#pragma unroll
      for (int h = 0; h < 16; ++h)
        bf[kk][h] = wq[(size_t)j * 128 + kk * 32 + frag_k(h, kg)];
    const float sc = (j < 128) ? QSCALE : 1.0f; // scale q third only
    for (int mt = 0; mt < 4; ++mt) {
      v8f acc = {};
#pragma unroll
      for (int kk = 0; kk < 4; ++kk) {
        v16h a;
#pragma unroll
        for (int h = 0; h < 16; ++h)
          a[h] = As[(mt * 16 + M) * 136 + kk * 32 + frag_k(h, kg)];
        WMMA_F32_F16(acc, a, bf[kk]);
      }
#pragma unroll
      for (int r = 0; r < 8; ++r) {             // C/D layout: row = r + 8*kg
        const int row = m0 + mt * 16 + r + 8 * kg;
        qkvh[(size_t)row * 384 + j] = (_Float16)(acc[r] * sc);
      }
    }
  }
}

// ---------------------------------------------------------------------------
// Kernel 2: per (window b, head h) fused attention.
//   TDM loads Q/K/V into padded LDS; S = QK^T (one K=32 WMMA per tile) with
//   bias gather + window mask fused at the D-layout store; in-place f16 row
//   softmax; O = P @ V (K padded to 128); f16 out to x2[b][n][h*32+d].
// ---------------------------------------------------------------------------
__global__ __launch_bounds__(256)
void attn_kernel(const _Float16* __restrict__ qkvh,
                 const float* __restrict__ mask,
                 const float* __restrict__ bias_table,
                 const int*   __restrict__ rel_index,
                 _Float16* __restrict__ x2) {
  __shared__ _Float16 Qs[112 * 40];   // 7x16 rows, 40-half padded stride
  __shared__ _Float16 Ks[112 * 40];
  __shared__ _Float16 Vs[128 * 40];   // m padded to 128 for the K-loop
  __shared__ _Float16 Sp[112 * 136];  // S, then P (in-place softmax)

  const int b   = blockIdx.x >> 2;
  const int h   = blockIdx.x & 3;
  const int tid = threadIdx.x;
  const int wid = tid >> 5, lane = tid & 31;
  const _Float16* src = qkvh + (size_t)b * NWIN * 384 + (size_t)h * HD;

  // --- Tensor Data Mover: stage Q/K/V (wave 0 issues; OOB rows zero-fill) ---
  if (wid == 0) {
    tdm_load_2d_f16((unsigned)(uintptr_t)&Qs[0], src,       32, NWIN, 112, 384, true);
    tdm_load_2d_f16((unsigned)(uintptr_t)&Ks[0], src + 128, 32, NWIN, 112, 384, true);
    tdm_load_2d_f16((unsigned)(uintptr_t)&Vs[0], src + 256, 32, NWIN, 128, 384, true);
    __builtin_amdgcn_s_wait_tensorcnt(0);
  }
  __syncthreads();

  const int L = lane & 15, kg = lane >> 4;
  const int w = b & 63;                          // window index: b % nW

  // ---- S = Q K^T : 49 tiles; bias + mask fused into the store ----
  for (int tile = wid; tile < 49; tile += 8) {
    const int qt = tile / 7, kt = tile % 7;
    v16h a, bf;
#pragma unroll
    for (int hh = 0; hh < 16; ++hh) {
      const int K = frag_k(hh, kg);
      a[hh]  = Qs[(qt * 16 + L) * 40 + K];       // row of Q (pre-scaled)
      bf[hh] = Ks[(kt * 16 + L) * 40 + K];       // row of K (B = K^T, same perm)
    }
    v8f acc = {};
    WMMA_F32_F16(acc, a, bf);
    const int m = kt * 16 + L;
#pragma unroll
    for (int r = 0; r < 8; ++r) {
      const int n = qt * 16 + r + 8 * kg;
      float v = acc[r];
      if (n < NWIN && m < NWIN)
        v += bias_table[rel_index[n * NWIN + m] * 4 + h] +
             mask[(size_t)w * (NWIN * NWIN) + (size_t)n * NWIN + m];
      Sp[n * 136 + m] = (_Float16)v;
    }
  }
  __syncthreads();

  // ---- row softmax, in place, f16 ----
  for (int n = tid; n < 112; n += 256) {
    _Float16* srow = Sp + n * 136;
    if (n < NWIN) {
      float mx = -1e30f;
      for (int m = 0; m < NWIN; ++m) mx = fmaxf(mx, (float)srow[m]);
      float sum = 0.f;
      for (int m = 0; m < NWIN; ++m) {
        const float e = __expf((float)srow[m] - mx);
        sum += e;
        srow[m] = (_Float16)e;
      }
      const float inv = 1.f / sum;
      for (int m = 0; m < NWIN; ++m) srow[m] = (_Float16)((float)srow[m] * inv);
      for (int m = NWIN; m < 128; ++m) srow[m] = (_Float16)0.f;
    } else {
      for (int m = 0; m < 128; ++m) srow[m] = (_Float16)0.f;
    }
  }
  __syncthreads();

  // ---- O = P @ V : 7 qt x 2 dt tiles, K = 128 (padded) in 4 steps ----
  for (int tile = wid; tile < 14; tile += 8) {
    const int qt = tile >> 1, dt = tile & 1;
    v8f acc = {};
#pragma unroll
    for (int kk = 0; kk < 4; ++kk) {
      v16h a, bf;
#pragma unroll
      for (int hh = 0; hh < 16; ++hh) {
        const int m = kk * 32 + frag_k(hh, kg);
        a[hh]  = Sp[(qt * 16 + L) * 136 + m];    // row of P
        bf[hh] = Vs[m * 40 + dt * 16 + L];       // column of V
      }
      WMMA_F32_F16(acc, a, bf);
    }
#pragma unroll
    for (int r = 0; r < 8; ++r) {
      const int n = qt * 16 + r + 8 * kg;
      if (n < NWIN)
        x2[((size_t)b * NWIN + n) * CDIM + h * HD + dt * 16 + L] =
            (_Float16)acc[r];
    }
  }
}

// ---------------------------------------------------------------------------
// Kernel 3: out = x2 @ proj_w^T + proj_b  (401408 x 128) @ (128 x 128), f32.
// ---------------------------------------------------------------------------
__global__ __launch_bounds__(256)
void proj_gemm(const _Float16* __restrict__ x2, const _Float16* __restrict__ wp,
               const float* __restrict__ proj_b, float* __restrict__ out) {
  __shared__ _Float16 As[64 * 136];
  const int m0  = blockIdx.x * 64;
  const int tid = threadIdx.x;
  for (int i = tid; i < 64 * 128; i += 256) {
    int r = i >> 7, c = i & 127;
    As[r * 136 + c] = x2[(size_t)(m0 + r) * 128 + c];
  }
  __syncthreads();

  const int wid = tid >> 5, lane = tid & 31;
  const int M = lane & 15, kg = lane >> 4;
  const int j = wid * 16 + M;                   // 8 waves x 16 cols = 128
  v16h bf[4];                                   // B fragments, reused over mt
#pragma unroll
  for (int kk = 0; kk < 4; ++kk)
#pragma unroll
    for (int h = 0; h < 16; ++h)
      bf[kk][h] = wp[(size_t)j * 128 + kk * 32 + frag_k(h, kg)];
  const float pb = proj_b[j];
  for (int mt = 0; mt < 4; ++mt) {
    v8f acc = {};
#pragma unroll
    for (int kk = 0; kk < 4; ++kk) {
      v16h a;
#pragma unroll
      for (int h = 0; h < 16; ++h)
        a[h] = As[(mt * 16 + M) * 136 + kk * 32 + frag_k(h, kg)];
      WMMA_F32_F16(acc, a, bf[kk]);
    }
#pragma unroll
    for (int r = 0; r < 8; ++r) {
      const int row = m0 + mt * 16 + r + 8 * kg;
      out[(size_t)row * 128 + j] = acc[r] + pb;
    }
  }
}

// ---------------------------------------------------------------------------
// Host entry
// ---------------------------------------------------------------------------
extern "C" void kernel_launch(void* const* d_in, const int* in_sizes, int n_in,
                              void* d_out, int out_size, void* d_ws,
                              size_t ws_size, hipStream_t stream) {
  (void)in_sizes; (void)n_in; (void)out_size; (void)ws_size;
  const float* x       = (const float*)d_in[0];
  const float* mask    = (const float*)d_in[1];
  const float* qkv_w   = (const float*)d_in[2];
  const float* proj_w  = (const float*)d_in[3];
  const float* proj_b  = (const float*)d_in[4];
  const float* bias_t  = (const float*)d_in[5];
  const int*   rel_idx = (const int*)d_in[6];
  float* out = (float*)d_out;

  // Workspace layout (all f16): qkv | x2 | wq | wp
  char* ws = (char*)d_ws;
  _Float16* qkvh = (_Float16*)ws;
  _Float16* x2   = (_Float16*)(ws + (size_t)ROWS * 384 * sizeof(_Float16));
  _Float16* wq   = (_Float16*)(ws + (size_t)ROWS * 384 * sizeof(_Float16) +
                               (size_t)ROWS * 128 * sizeof(_Float16));
  _Float16* wp   = wq + 384 * 128;

  convert_weights<<<(384 * 128 + 255) / 256, 256, 0, stream>>>(qkv_w, proj_w,
                                                               wq, wp);
  qkv_gemm<<<ROWS / 64, 256, 0, stream>>>(x, wq, qkvh);
  attn_kernel<<<4096 * HEADS, 256, 0, stream>>>(qkvh, mask, bias_t, rel_idx, x2);
  proj_gemm<<<ROWS / 64, 256, 0, stream>>>(x2, wp, proj_b, out);
}
